// ACPD_73443940762333
// MI455X (gfx1250) — compile-verified, matrix operand order checked by reading, not run on hardware
//
#include <hip/hip_runtime.h>
#include <stdint.h>

// ---- problem constants (from reference) ----
#define BB      8
#define HH      128
#define WW      128
#define CC      64
#define FF      64
#define HW      (HH*WW)       // 16384
#define KDIM    576           // 9 * 64  (im2col K)
#define MTILE   64            // output pixels per workgroup
#define THREADS 256           // 8 wave32s

typedef __attribute__((ext_vector_type(16))) __bf16 v16bf;
typedef __attribute__((ext_vector_type(8)))  float  v8f;

union FragU {
    uint4  q[2];   // two ds_load_b128 halves
    v16bf  v;
};

__device__ __forceinline__ unsigned short f2bf(float x) {
    union { float f; uint32_t u; } v; v.f = x;
    uint32_t r = v.u + 0x7FFFu + ((v.u >> 16) & 1u);   // round-to-nearest-even
    return (unsigned short)(r >> 16);
}

// sum of the two largest of {a,b,c,d}, each clipped to [1, 127]
__device__ __forceinline__ float top2clipsum(float a, float b, float c, float d) {
    float h1 = fmaxf(a, b), l1 = fminf(a, b);
    float h2 = fmaxf(c, d), l2 = fminf(c, d);
    float m1 = fmaxf(h1, h2);
    float m2 = fmaxf(fminf(h1, h2), fmaxf(l1, l2));
    m1 = fminf(fmaxf(m1, 1.0f), 127.0f);
    m2 = fminf(fmaxf(m2, 1.0f), 127.0f);
    return m1 + m2;
}

__global__ __launch_bounds__(THREADS)
void acpd_fused_wmma(const float* __restrict__ fmap,
                     const float* __restrict__ y2,
                     const float* __restrict__ conv_w,
                     const float* __restrict__ gamma,
                     const float* __restrict__ beta,
                     const float* __restrict__ mmean,
                     const float* __restrict__ mvar,
                     float* __restrict__ out)
{
    extern __shared__ __align__(16) unsigned short smem[];
    unsigned short* Wt  = smem;             // [FF][KDIM] bf16, weights transposed (N-major)
    unsigned short* Ash = smem + FF*KDIM;   // [MTILE][KDIM] bf16, im2col A-tile

    const int tid   = threadIdx.x;
    const int mbase = blockIdx.x * MTILE;   // global output-pixel row base

    // ---- stage weights: conv_w[(k,f)] (f32, k = (di*3+dj)*64+c) -> Wt[f][k] (bf16) ----
    for (int idx = tid; idx < KDIM*FF; idx += THREADS) {
        int k = idx >> 6;      // /64
        int f = idx & 63;
        Wt[f*KDIM + k] = f2bf(conv_w[idx]);
    }

    // ---- sampling phase: build A-tile (64 rows x 576) ----
    // 576 tasks (pixel,tap) x 8 channel-groups of 8 channels = 4608 units / 256 thr = 18 iters
    #pragma unroll 1
    for (int it = 0; it < (MTILE*9*8)/THREADS; ++it) {
        int u    = it*THREADS + tid;
        int task = u >> 3;
        int c0   = (u & 7) << 3;
        int p    = task / 9;            // local pixel 0..63
        int d    = task - p*9;          // tap 0..8
        int di   = d / 3, dj = d - di*3;

        int g   = mbase + p;            // global pixel
        int bb  = g >> 14;              // / 16384
        int rem = g & 16383;
        int oi  = rem >> 7;
        int oj  = rem & 127;

        // n = 1152*oi + 384*di + 3*oj + dj ;  hw = n/9 ; kk = (n%9)/3
        int oj3 = oj / 3;
        int r   = 6*di + 3*(oj - oj3*3) + dj;       // 0..20
        int r9  = r / 9;
        int hw  = 128*oi + 42*di + oj3 + r9;        // source pixel
        int kk  = (r - r9*9) / 3;                   // variant 0..2 (x and y)
        int pi  = hw >> 7, pj = hw & 127;

        const float* yp = y2 + ((size_t)bb*HW + hw)*8;
        float wv = top2clipsum(yp[0], yp[1], yp[2], yp[3]);
        float hv = top2clipsum(yp[4], yp[5], yp[6], yp[7]);

        float xv, yv;
        if (kk == 0)      { xv = (float)pj - 1.0f - wv*(1.0f/3.0f); yv = (float)pi - 1.0f - hv*(1.0f/3.0f); }
        else if (kk == 1) { xv = (float)pj + wv*1e-10f;             yv = (float)pi + hv*1e-10f; }
        else              { xv = (float)pj + 1.0f + wv*(1.0f/3.0f); yv = (float)pi + 1.0f + hv*(1.0f/3.0f); }

        float xl = fminf(fmaxf(floorf(xv), 0.0f), 127.0f);
        float xr = fminf(fmaxf(ceilf(xv),  0.0f), 127.0f);
        float yt = fminf(fmaxf(floorf(yv), 0.0f), 127.0f);
        float yb = fminf(fmaxf(ceilf(yv),  0.0f), 127.0f);
        float wxr = xr - xv, wxl = xv - xl;         // reference's (clipped) weights
        float wyt = yv - yt, wyb = yb - yv;
        int ixl = (int)xl, ixr = (int)xr, iyt = (int)yt, iyb = (int)yb;

        const float* fb  = fmap + (size_t)bb*HW*CC;
        const float* plt = fb + ((size_t)(iyt*WW + ixl))*CC + c0;
        const float* prt = fb + ((size_t)(iyt*WW + ixr))*CC + c0;
        const float* plb = fb + ((size_t)(iyb*WW + ixl))*CC + c0;
        const float* prb = fb + ((size_t)(iyb*WW + ixr))*CC + c0;

        unsigned short* arow = Ash + p*KDIM + d*CC + c0;
        #pragma unroll
        for (int c = 0; c < 8; ++c) {
            // f = (wxr*f_lt + wxl*f_lb)*wyb + (wxr*f_rt + wxl*f_rb)*wyt  (reference formula)
            float v = (wxr*plt[c] + wxl*plb[c])*wyb + (wxr*prt[c] + wxl*prb[c])*wyt;
            arow[c] = f2bf(v);
        }
    }
    __syncthreads();

    // ---- WMMA phase: 8 waves = 4 (M) x 2 (N-halves); each wave: 16x32 output ----
    const int wave = tid >> 5;
    const int lane = tid & 31;
    const int m0   = (wave & 3) * 16;
    const int n0   = (wave >> 2) * 32;
    const int rsel = lane & 15;            // M-row (A) / N-col (B) within tile
    const int kofs = (lane >> 4) << 3;     // K 0..7 vs 8..15 halves per ISA layout

    const unsigned short* Abase  = Ash + (m0 + rsel)*KDIM + kofs;
    const unsigned short* B0base = Wt  + (n0 + rsel)*KDIM + kofs;
    const unsigned short* B1base = Wt  + (n0 + 16 + rsel)*KDIM + kofs;

    v8f acc0 = {}; v8f acc1 = {};
    #pragma unroll 3
    for (int kc = 0; kc < KDIM; kc += 32) {
        FragU a, b0, b1;
        a.q[0]  = *(const uint4*)(Abase  + kc);
        a.q[1]  = *(const uint4*)(Abase  + kc + 16);
        b0.q[0] = *(const uint4*)(B0base + kc);
        b0.q[1] = *(const uint4*)(B0base + kc + 16);
        b1.q[0] = *(const uint4*)(B1base + kc);
        b1.q[1] = *(const uint4*)(B1base + kc + 16);
        acc0 = __builtin_amdgcn_wmma_f32_16x16x32_bf16(false, a.v, false, b0.v,
                                                       (short)0, acc0, false, false);
        acc1 = __builtin_amdgcn_wmma_f32_16x16x32_bf16(false, a.v, false, b1.v,
                                                       (short)0, acc1, false, false);
    }

    // ---- BN + ReLU epilogue (C/D layout: VGPR r -> M = r + 8*(lane>=16), N = lane%16) ----
    const int colA = n0 + rsel;
    const int colB = colA + 16;
    float sA = gamma[colA] * rsqrtf(mvar[colA] + 1e-3f);
    float sB = gamma[colB] * rsqrtf(mvar[colB] + 1e-3f);
    float mA = mmean[colA], bA = beta[colA];
    float mB = mmean[colB], bB = beta[colB];
    const int rowoff = (lane >> 4) << 3;

    #pragma unroll
    for (int rr = 0; rr < 8; ++rr) {
        size_t orow = (size_t)(mbase + m0 + rowoff + rr) * FF;
        out[orow + colA] = fmaxf(0.0f, (acc0[rr] - mA)*sA + bA);
        out[orow + colB] = fmaxf(0.0f, (acc1[rr] - mB)*sB + bB);
    }
}

extern "C" void kernel_launch(void* const* d_in, const int* in_sizes, int n_in,
                              void* d_out, int out_size, void* d_ws, size_t ws_size,
                              hipStream_t stream) {
    (void)in_sizes; (void)n_in; (void)out_size; (void)d_ws; (void)ws_size;
    const float* fmap   = (const float*)d_in[0];
    const float* y2     = (const float*)d_in[1];
    const float* conv_w = (const float*)d_in[2];
    const float* gamma  = (const float*)d_in[3];
    const float* beta   = (const float*)d_in[4];
    const float* mmean  = (const float*)d_in[5];
    const float* mvar   = (const float*)d_in[6];
    float* out          = (float*)d_out;

    const int blocks = (BB*HW) / MTILE;                                  // 2048
    const size_t shmem = (size_t)(FF + MTILE) * KDIM * sizeof(unsigned short); // 144 KB (< 320 KB WGP LDS)

    acpd_fused_wmma<<<blocks, THREADS, shmem, stream>>>(
        fmap, y2, conv_w, gamma, beta, mmean, mvar, out);
}
